// SelfAttention_3109556322913
// MI455X (gfx1250) — matrix operand order repaired and checked by reading, not compiled, not run on hardware
//
#include <hip/hip_runtime.h>
#include <hip/hip_bf16.h>

// ---------------------------------------------------------------------------
// Self-attention, CDNA5/gfx1250, f16 WMMA pipeline (f32 accumulate).
// N=4, L=2048, E=1024, H=16, D=64.
// ---------------------------------------------------------------------------

typedef __attribute__((ext_vector_type(16))) _Float16 v16h;
typedef __attribute__((ext_vector_type(8)))  float    v8f;

#define WMMA16(a, b, c) \
  __builtin_amdgcn_wmma_f32_16x16x32_f16(false, (a), false, (b), (short)0, (c), false, false)

static constexpr int N_ = 4;
static constexpr int L_ = 2048;
static constexpr int E_ = 1024;
static constexpr int H_ = 16;
static constexpr int D_ = 64;

// ---- WMMA fragment loaders (layouts per CDNA5 ISA 7.12.2, wave32) ----------
// A (16xK=32 f16): lane = M row (lane&15); VGPR v, half j:
//   v<4 : K = 2v + 8g + j       v>=4 : K = 16 + 2(v-4) + 8g + j   (g = lane>>4)
__device__ __forceinline__ v16h load_a_frag(const _Float16* p, int ld) {
  const int lane = threadIdx.x & 31;
  const int r = lane & 15, g = lane >> 4;
  v16h a;
#pragma unroll
  for (int v = 0; v < 8; ++v) {
    const int kb = (v < 4) ? (2 * v + 8 * g) : (16 + 2 * (v - 4) + 8 * g);
    a[2 * v]     = p[r * ld + kb];
    a[2 * v + 1] = p[r * ld + kb + 1];
  }
  return a;
}

// B (K=32 x 16 f16) loaded from a transposed (Bt[n][k], row-major) tile:
// lane = N column (lane&15); VGPR v, half j: same K as load_a_frag.
__device__ __forceinline__ v16h load_bt_frag(const _Float16* p, int ld) {
  const int lane = threadIdx.x & 31;
  const int n = lane & 15, g = lane >> 4;
  v16h b;
#pragma unroll
  for (int v = 0; v < 8; ++v) {
    const int kb = (v < 4) ? (2 * v + 8 * g) : (16 + 2 * (v - 4) + 8 * g);
    b[2 * v]     = p[n * ld + kb];
    b[2 * v + 1] = p[n * ld + kb + 1];
  }
  return b;
}

// ---------------------------------------------------------------------------
// Kernel 0: Wo f32 -> f16 (row-major [e_out][e_in], matches Bt layout).
// ---------------------------------------------------------------------------
__global__ __launch_bounds__(256) void wo_convert_kernel(const float* __restrict__ Wo,
                                                         _Float16* __restrict__ Wo16) {
  int idx = blockIdx.x * 256 + threadIdx.x;
  const int total = E_ * E_;
  for (int i = idx; i < total; i += gridDim.x * 256) Wo16[i] = (_Float16)Wo[i];
}

// ---------------------------------------------------------------------------
// Kernel 1: per-head projections.  Each wave handles one (n,l): 16 rows (h) x
// 64 cols (d).  Writes Qp/Kp [n][h][l][d] f16 and V transposed [n][h][d][l].
// ---------------------------------------------------------------------------
__global__ __launch_bounds__(128) void proj_kernel(
    const float* __restrict__ q, const float* __restrict__ k, const float* __restrict__ v,
    const float* __restrict__ Wq, const float* __restrict__ Wk, const float* __restrict__ Wv,
    _Float16* __restrict__ Qp, _Float16* __restrict__ Kp, _Float16* __restrict__ Vt) {
  __shared__ _Float16 w_lds[3][D_ * D_];          // 24 KB, W[e_out][d_in] = Bt layout
  __shared__ _Float16 x_lds[4][3][16 * D_];       // 24 KB, per-wave X tiles

  const int t = threadIdx.x, wave = t >> 5, lane = t & 31;
  for (int i = t; i < D_ * D_; i += 128) {
    w_lds[0][i] = (_Float16)Wq[i];
    w_lds[1][i] = (_Float16)Wk[i];
    w_lds[2][i] = (_Float16)Wv[i];
  }

  const long pair = (long)blockIdx.x * 4 + wave;  // flat (n*L + l)
  const int n = (int)(pair / L_), l = (int)(pair % L_);
  const float* srcs[3] = {q, k, v};
#pragma unroll
  for (int m = 0; m < 3; ++m) {
    const float* src = srcs[m] + pair * E_;
    for (int i = lane; i < E_; i += 32) x_lds[wave][m][i] = (_Float16)src[i];
  }
  __syncthreads();

  const int r = lane & 15, g = lane >> 4;
#pragma unroll
  for (int m = 0; m < 3; ++m) {
    const v16h a0 = load_a_frag(&x_lds[wave][m][0], D_);
    const v16h a1 = load_a_frag(&x_lds[wave][m][32], D_);
#pragma unroll
    for (int ec = 0; ec < 4; ++ec) {
      v8f c = {};
      c = WMMA16(a0, load_bt_frag(&w_lds[m][(ec * 16) * D_ + 0], D_), c);
      c = WMMA16(a1, load_bt_frag(&w_lds[m][(ec * 16) * D_ + 32], D_), c);
      const int col = ec * 16 + r;
#pragma unroll
      for (int vv = 0; vv < 8; ++vv) {
        const int h = vv + 8 * g;                 // C row M -> head index
        const _Float16 val = (_Float16)c[vv];
        if (m == 0)      Qp[(((long)n * H_ + h) * L_ + l) * D_ + col] = val;
        else if (m == 1) Kp[(((long)n * H_ + h) * L_ + l) * D_ + col] = val;
        else             Vt[(((long)n * H_ + h) * D_ + col) * (long)L_ + l] = val;
      }
    }
  }
}

// ---------------------------------------------------------------------------
// Kernel 2: flash attention.  Block = 4 waves = 64 q rows of one (n,h); each
// wave owns a 16-row q tile with online softmax; 32-key chunks through LDS.
// ---------------------------------------------------------------------------
__global__ __launch_bounds__(128) void attn_kernel(
    const _Float16* __restrict__ Qp, const _Float16* __restrict__ Kp,
    const _Float16* __restrict__ Vt, const int* __restrict__ mask,
    _Float16* __restrict__ Of16) {
  __shared__ _Float16 kt_lds[32 * D_];            // K tile [key][d]   (4 KB)
  __shared__ _Float16 vt_lds[D_ * 32];            // V^T tile [d][key] (4 KB)
  __shared__ _Float16 p_lds[4][16 * 32];          // per-wave P scratch (4 KB)

  const int t = threadIdx.x, wave = t >> 5, lane = t & 31;
  const int r = lane & 15, g = lane >> 4;

  const int b = blockIdx.x;
  const int qt_per = L_ / 64;
  const int n = b / (H_ * qt_per);
  const int rem = b % (H_ * qt_per);
  const int h = rem / qt_per;
  const int q0 = (rem % qt_per) * 64 + wave * 16;

  const _Float16* Qh = Qp + ((long)n * H_ + h) * (long)L_ * D_;
  const _Float16* Kh = Kp + ((long)n * H_ + h) * (long)L_ * D_;
  const _Float16* Vh = Vt + ((long)n * H_ + h) * (long)D_ * L_;

  const v16h qa0 = load_a_frag(Qh + (long)q0 * D_, D_);
  const v16h qa1 = load_a_frag(Qh + (long)q0 * D_ + 32, D_);

  v8f o0 = {}, o1 = {}, o2 = {}, o3 = {};
  float mrow[8], lrow[8];
#pragma unroll
  for (int i = 0; i < 8; ++i) { mrow[i] = -1e30f; lrow[i] = 0.f; }

  const float scale = 0.03125f;                   // 1/sqrt(E) = 1/32

  for (int kt = 0; kt < L_ / 32; ++kt) {
    const int k0 = kt * 32;
    __syncthreads();                              // previous tiles consumed
    {                                             // cooperative tile loads
      const uint4* ksrc = (const uint4*)(Kh + (long)k0 * D_);   // contiguous 32x64
      uint4* kdst = (uint4*)kt_lds;
      for (int i = t; i < 256; i += 128) kdst[i] = ksrc[i];
      uint4* vdst = (uint4*)vt_lds;
      for (int i = t; i < 256; i += 128) {
        const int row = i >> 2, c4 = i & 3;       // 64 rows x 4 uint4
        vdst[i] = *(const uint4*)(Vh + (long)row * L_ + k0 + c4 * 8);
      }
      if (kt + 1 < L_ / 32) {                     // global_prefetch_b8 hints
        __builtin_prefetch(Kh + (long)(k0 + 32) * D_, 0, 0);
        __builtin_prefetch(Vh + k0 + 32, 0, 0);
      }
    }
    __syncthreads();

    // S = Q K^T tile: 16 q x 32 keys, f32 accumulate
    v8f s0 = {}, s1 = {};
    s0 = WMMA16(qa0, load_bt_frag(kt_lds, D_), s0);
    s0 = WMMA16(qa1, load_bt_frag(kt_lds + 32, D_), s0);
    s1 = WMMA16(qa0, load_bt_frag(kt_lds + 16 * D_, D_), s1);
    s1 = WMMA16(qa1, load_bt_frag(kt_lds + 16 * D_ + 32, D_), s1);

    const int* mbase = mask + ((long)n * L_ + q0) * L_ + k0;
    float x0[8], x1[8], rmax[8];
#pragma unroll
    for (int vv = 0; vv < 8; ++vv) {
      const int qq = vv + 8 * g;                  // C row of this VGPR
      float a = s0[vv] * scale;
      float c = s1[vv] * scale;
      if (mbase[(long)qq * L_ + r] == 0)      a = -1e30f;
      if (mbase[(long)qq * L_ + 16 + r] == 0) c = -1e30f;
      x0[vv] = a; x1[vv] = c;
      float mx = fmaxf(a, c);                     // row-max across 16 lanes
#pragma unroll
      for (int off = 1; off < 16; off <<= 1) mx = fmaxf(mx, __shfl_xor(mx, off, 32));
      rmax[vv] = mx;
    }

#pragma unroll
    for (int vv = 0; vv < 8; ++vv) {
      const float mnew = fmaxf(mrow[vv], rmax[vv]);
      const float alpha = __expf(mrow[vv] - mnew);
      const float e0 = __expf(x0[vv] - mnew);
      const float e1 = __expf(x1[vv] - mnew);
      float rs = e0 + e1;
#pragma unroll
      for (int off = 1; off < 16; off <<= 1) rs += __shfl_xor(rs, off, 32);
      lrow[vv] = lrow[vv] * alpha + rs;
      mrow[vv] = mnew;
      o0[vv] *= alpha; o1[vv] *= alpha; o2[vv] *= alpha; o3[vv] *= alpha;
      _Float16* pw = &p_lds[wave][(vv + 8 * g) * 32];
      pw[r]      = (_Float16)e0;
      pw[16 + r] = (_Float16)e1;
    }
    __syncthreads();                              // P visible across lanes

    const v16h pa = load_a_frag(&p_lds[wave][0], 32);   // 16x32 A fragment
    o0 = WMMA16(pa, load_bt_frag(vt_lds + 0 * 32, 32), o0);
    o1 = WMMA16(pa, load_bt_frag(vt_lds + 16 * 32, 32), o1);
    o2 = WMMA16(pa, load_bt_frag(vt_lds + 32 * 32, 32), o2);
    o3 = WMMA16(pa, load_bt_frag(vt_lds + 48 * 32, 32), o3);
  }

#pragma unroll
  for (int vv = 0; vv < 8; ++vv) {
    const float inv = 1.0f / lrow[vv];
    const long qq = q0 + vv + 8 * g;
    const long base = ((long)n * L_ + qq) * E_ + (long)h * D_;
    Of16[base + r]      = (_Float16)(o0[vv] * inv);
    Of16[base + 16 + r] = (_Float16)(o1[vv] * inv);
    Of16[base + 32 + r] = (_Float16)(o2[vv] * inv);
    Of16[base + 48 + r] = (_Float16)(o3[vv] * inv);
  }
}

// ---------------------------------------------------------------------------
// Kernel 3: out = O @ Wo^T + bo.  64x64 output tile per block, K=1024.
// ---------------------------------------------------------------------------
__global__ __launch_bounds__(128) void outproj_kernel(
    const _Float16* __restrict__ A, const _Float16* __restrict__ B,
    const float* __restrict__ bias, float* __restrict__ out) {
  __shared__ _Float16 a_lds[64 * 32];             // O rows
  __shared__ _Float16 b_lds[64 * 32];             // Wo rows (Bt layout)
  const int t = threadIdx.x, wave = t >> 5, lane = t & 31;
  const int rt = blockIdx.x % (N_ * L_ / 64);
  const int ct = blockIdx.x / (N_ * L_ / 64);
  const long row0 = (long)rt * 64;
  const int col0 = ct * 64;

  v8f c0 = {}, c1 = {}, c2 = {}, c3 = {};
  for (int kc = 0; kc < E_ / 32; ++kc) {
    __syncthreads();
    for (int i = t; i < 256; i += 128) {          // 64 rows x 4 uint4 each tile
      const int rr = i >> 2, cc = i & 3;
      ((uint4*)a_lds)[i] = *(const uint4*)(A + (row0 + rr) * (long)E_ + kc * 32 + cc * 8);
      ((uint4*)b_lds)[i] = *(const uint4*)(B + (long)(col0 + rr) * E_ + kc * 32 + cc * 8);
    }
    __syncthreads();
    const v16h af = load_a_frag(a_lds + wave * 16 * 32, 32);
    c0 = WMMA16(af, load_bt_frag(b_lds + 0 * 32, 32), c0);
    c1 = WMMA16(af, load_bt_frag(b_lds + 16 * 32, 32), c1);
    c2 = WMMA16(af, load_bt_frag(b_lds + 32 * 32, 32), c2);
    c3 = WMMA16(af, load_bt_frag(b_lds + 48 * 32, 32), c3);
  }

  const int r = lane & 15, g = lane >> 4;
#pragma unroll
  for (int vv = 0; vv < 8; ++vv) {
    const long row = row0 + wave * 16 + vv + 8 * g;
    float* o = out + row * E_ + col0;
    o[r]      = c0[vv] + bias[col0 + r];
    o[16 + r] = c1[vv] + bias[col0 + 16 + r];
    o[32 + r] = c2[vv] + bias[col0 + 32 + r];
    o[48 + r] = c3[vv] + bias[col0 + 48 + r];
  }
}

// ---------------------------------------------------------------------------
extern "C" void kernel_launch(void* const* d_in, const int* in_sizes, int n_in,
                              void* d_out, int out_size, void* d_ws, size_t ws_size,
                              hipStream_t stream) {
  const float* q   = (const float*)d_in[0];
  const float* k   = (const float*)d_in[1];
  const float* v   = (const float*)d_in[2];
  const int*   msk = (const int*)d_in[3];
  const float* Wq  = (const float*)d_in[4];
  const float* Wk  = (const float*)d_in[5];
  const float* Wv  = (const float*)d_in[6];
  const float* Wo  = (const float*)d_in[7];
  const float* bo  = (const float*)d_in[8];
  float* out = (float*)d_out;

  // Workspace layout (f16): Qp/Kp/Vt/Of16 16 MB each + Wo16 2 MB = 66 MB.
  const size_t seg = (size_t)N_ * H_ * L_ * D_ * sizeof(_Float16);  // 16 MB
  char* ws = (char*)d_ws;
  _Float16* Qp   = (_Float16*)(ws);
  _Float16* Kp   = (_Float16*)(ws + seg);
  _Float16* Vt   = (_Float16*)(ws + 2 * seg);
  _Float16* Of16 = (_Float16*)(ws + 3 * seg);
  _Float16* Wo16 = (_Float16*)(ws + 4 * seg);

  wo_convert_kernel<<<1024, 256, 0, stream>>>(Wo, Wo16);
  proj_kernel<<<(N_ * L_) / 4, 128, 0, stream>>>(q, k, v, Wq, Wk, Wv, Qp, Kp, Vt);
  attn_kernel<<<N_ * H_ * (L_ / 64), 128, 0, stream>>>(Qp, Kp, Vt, msk, Of16);
  outproj_kernel<<<(N_ * L_ / 64) * (E_ / 64), 128, 0, stream>>>(Of16, Wo16, bo, out);
}
